// EncoderLayer_55310588837917
// MI455X (gfx1250) — matrix operand order repaired and checked by reading, not compiled, not run on hardware
//
#include <hip/hip_runtime.h>
#include <hip/hip_bf16.h>
#include <stdint.h>

typedef __bf16 bf16_t;
typedef __attribute__((ext_vector_type(16))) __bf16 v16bf;
typedef __attribute__((ext_vector_type(8)))  float v8f;

#define WMMA_BF16(a, b, c) \
  __builtin_amdgcn_wmma_f32_16x16x32_bf16(false, (a), false, (b), (short)0, (c), false, false)

union FragBF { v16bf v; uint4 q[2]; uint32_t u[8]; };

static constexpr int Bsz = 4, Sseq = 2048, Dmod = 1024, Hh = 16, DFFf = 4096;
static constexpr int NT = Bsz * Sseq;          // 8192 tokens
static constexpr int NQKV = 3 * Dmod;          // 3072

// ---------------------------------------------------------------------------
// CDNA5 async global->LDS copy (16B per lane), tracked by ASYNCcnt.
// GV mode: global_load_async_to_lds_b128 vdst(LDS addr), vaddr(64b), off
// ---------------------------------------------------------------------------
__device__ __forceinline__ void async_copy16(bf16_t* lds, const bf16_t* g) {
  uint32_t l = (uint32_t)(uintptr_t)lds;     // low 32 bits of flat LDS addr = LDS offset
  uint64_t ga = (uint64_t)(uintptr_t)g;
  asm volatile("global_load_async_to_lds_b128 %0, %1, off"
               :: "v"(l), "v"(ga) : "memory");
}
__device__ __forceinline__ void wait_async0() {
  asm volatile("s_wait_asynccnt 0x0" ::: "memory");
}

// ---------------------------------------------------------------------------
// Weight conversion: f32 [K,N] -> bf16 [N,K] (transposed so WMMA B-fragment
// dword pairs are contiguous)
// ---------------------------------------------------------------------------
__global__ __launch_bounds__(256) void transpose_f32_to_bf16(
    const float* __restrict__ W, bf16_t* __restrict__ Wt, int K, int N) {
  int idx = blockIdx.x * 256 + threadIdx.x;
  if (idx >= N * K) return;
  int n = idx / K, k = idx - n * K;
  Wt[idx] = (bf16_t)W[(size_t)k * N + n];
}

// Build combined QKV weight: Wt[3072][1024], output col c = part*1024 + h*64 + dk
__global__ __launch_bounds__(256) void build_wqkvt(
    const float* __restrict__ Wq, const float* __restrict__ Wk,
    const float* __restrict__ Wv, bf16_t* __restrict__ Wt) {
  int idx = blockIdx.x * 256 + threadIdx.x;    // over 3072*1024
  int c = idx >> 10, d = idx & 1023;
  const float* W = (c < 1024) ? Wq : (c < 2048) ? Wk : Wv;
  int cc = c & 1023;
  // W[h, d, dk] with strides (65536, 64, 1)
  Wt[idx] = (bf16_t)W[(size_t)(cc >> 6) * 65536 + (size_t)d * 64 + (cc & 63)];
}

// ---------------------------------------------------------------------------
// RMSNorm: f32 [rows,1024] -> bf16 [rows,1024]
// ---------------------------------------------------------------------------
__global__ __launch_bounds__(256) void rmsnorm_bf16_kernel(
    const float* __restrict__ x, const float* __restrict__ g,
    bf16_t* __restrict__ out) {
  const int row = blockIdx.x, tid = threadIdx.x;
  const float* xr = x + (size_t)row * Dmod;
  float v0 = xr[tid], v1 = xr[tid + 256], v2 = xr[tid + 512], v3 = xr[tid + 768];
  float s = v0 * v0 + v1 * v1 + v2 * v2 + v3 * v3;
  for (int off = 16; off; off >>= 1) s += __shfl_xor(s, off, 32);
  __shared__ float red[8];
  int wave = tid >> 5, lane = tid & 31;
  if (lane == 0) red[wave] = s;
  __syncthreads();
  float tot = red[0] + red[1] + red[2] + red[3] + red[4] + red[5] + red[6] + red[7];
  float inv = rsqrtf(tot * (1.0f / (float)Dmod));
  bf16_t* o = out + (size_t)row * Dmod;
  o[tid]       = (bf16_t)(g[tid] * v0 * inv);
  o[tid + 256] = (bf16_t)(g[tid + 256] * v1 * inv);
  o[tid + 512] = (bf16_t)(g[tid + 512] * v2 * inv);
  o[tid + 768] = (bf16_t)(g[tid + 768] * v3 * inv);
}

// ---------------------------------------------------------------------------
// Tiled bf16 GEMM, f32 accumulate via v_wmma_f32_16x16x32_bf16.
//   A: [M,K] row-major bf16 (lda)      Bt: [N,K] row-major bf16 (ldb)
//   EPI 1: C.f32 = addend + acc        EPI 2: C.bf16 = acc
//   EPI 3: C.bf16 = silu(acc) * acc2 (dual B: Bt=W1t, B2t=V1t)
// Macro tile 128x128, BK=32, 256 threads = 8 waves (4 Mwaves x 2 Nwaves),
// wave tile 32x64 = 2x4 WMMA tiles, double-buffered LDS with async DMA.
// ---------------------------------------------------------------------------
template <int EPI>
__global__ __launch_bounds__(256) void gemm_bf16_kernel(
    const bf16_t* __restrict__ A, int lda,
    const bf16_t* __restrict__ Bt, const bf16_t* __restrict__ B2t, int ldb,
    const float* __restrict__ addend, void* __restrict__ Cptr, int ldc,
    int M, int N, int K) {
  constexpr int LDSROW = 40;  // elements (80B row, 16B aligned, bank-skewed)
  __shared__ __align__(16) bf16_t sA[2][128 * LDSROW];
  __shared__ __align__(16) bf16_t sB[2][128 * LDSROW];
  __shared__ __align__(16) bf16_t sB2[(EPI == 3) ? 2 : 1][(EPI == 3) ? 128 * LDSROW : 1];

  const int tid = threadIdx.x;
  const int lane = tid & 31, wave = tid >> 5;
  const int lh = lane >> 4, lm = lane & 15;
  const int wm = wave >> 1;  // 0..3 -> 32 rows each
  const int wn = wave & 1;   // 0..1 -> 64 cols each
  const int m0 = blockIdx.y * 128;
  const int n0 = blockIdx.x * 128;

  v8f acc[2][4];
  v8f acc2[(EPI == 3) ? 2 : 1][(EPI == 3) ? 4 : 1];
  for (int i = 0; i < 2; ++i)
    for (int j = 0; j < 4; ++j) acc[i][j] = {};
  if constexpr (EPI == 3)
    for (int i = 0; i < 2; ++i)
      for (int j = 0; j < 4; ++j) acc2[i][j] = {};

  const int nk = K >> 5;
  auto load_tiles = [&](int kt, int buf) {
    const int kb = kt << 5;
    for (int i = 0; i < 2; ++i) {
      int c = tid + (i << 8);
      int row = c >> 2, off8 = (c & 3) << 3;
      const bf16_t* ga = A + (size_t)(m0 + row) * lda + kb + off8;
      const bf16_t* gb = Bt + (size_t)(n0 + row) * ldb + kb + off8;
      async_copy16(&sA[buf][row * LDSROW + off8], ga);
      async_copy16(&sB[buf][row * LDSROW + off8], gb);
      if (kb + 64 <= K) {  // warm L2 for the tile after next
        __builtin_prefetch(ga + 32, 0, 0);
        __builtin_prefetch(gb + 32, 0, 0);
      }
      if constexpr (EPI == 3)
        async_copy16(&sB2[buf][row * LDSROW + off8],
                     B2t + (size_t)(n0 + row) * ldb + kb + off8);
    }
  };

  load_tiles(0, 0);
  wait_async0();
  __syncthreads();
  int buf = 0;
  for (int kt = 0; kt < nk; ++kt) {
    if (kt + 1 < nk) load_tiles(kt + 1, buf ^ 1);  // DMA overlaps the WMMAs below
    FragBF af[2], bfrag[4], bfrag2[4];
    for (int mt = 0; mt < 2; ++mt) {
      const uint32_t* base =
          (const uint32_t*)&sA[buf][(wm * 32 + mt * 16 + lm) * LDSROW];
      af[mt].q[0] = *(const uint4*)(base + lh * 4);       // K = lh*8 + {0..7}
      af[mt].q[1] = *(const uint4*)(base + 8 + lh * 4);   // K = 16 + lh*8 + {0..7}
    }
    for (int nt = 0; nt < 4; ++nt) {
      const uint32_t* base =
          (const uint32_t*)&sB[buf][(wn * 64 + nt * 16 + lm) * LDSROW];
      bfrag[nt].q[0] = *(const uint4*)(base + lh * 4);     // K = lh*8 + {0..7}
      bfrag[nt].q[1] = *(const uint4*)(base + 8 + lh * 4); // K = 16 + lh*8 + {0..7}
      if constexpr (EPI == 3) {
        const uint32_t* b2 =
            (const uint32_t*)&sB2[buf][(wn * 64 + nt * 16 + lm) * LDSROW];
        bfrag2[nt].q[0] = *(const uint4*)(b2 + lh * 4);
        bfrag2[nt].q[1] = *(const uint4*)(b2 + 8 + lh * 4);
      }
    }
    for (int mt = 0; mt < 2; ++mt)
      for (int nt = 0; nt < 4; ++nt) {
        acc[mt][nt] = WMMA_BF16(af[mt].v, bfrag[nt].v, acc[mt][nt]);
        if constexpr (EPI == 3)
          acc2[mt][nt] = WMMA_BF16(af[mt].v, bfrag2[nt].v, acc2[mt][nt]);
      }
    wait_async0();
    __syncthreads();
    buf ^= 1;
  }

  for (int mt = 0; mt < 2; ++mt)
    for (int nt = 0; nt < 4; ++nt)
      for (int r = 0; r < 8; ++r) {
        int row = m0 + wm * 32 + mt * 16 + r + lh * 8;  // C layout: M = r + 8*half
        int col = n0 + wn * 64 + nt * 16 + lm;          // N = lane%16
        size_t idx = (size_t)row * ldc + col;
        float v = acc[mt][nt][r];
        if constexpr (EPI == 1) {
          ((float*)Cptr)[idx] = addend[idx] + v;
        } else if constexpr (EPI == 2) {
          ((bf16_t*)Cptr)[idx] = (bf16_t)v;
        } else {
          float v2 = acc2[mt][nt][r];
          float sil = v / (1.0f + __expf(-v));
          ((bf16_t*)Cptr)[idx] = (bf16_t)(sil * v2);
        }
      }
}

// ---------------------------------------------------------------------------
// Flash attention, one (b,h) per blockIdx.y, 64 query rows per block,
// 128 threads = 4 waves, each wave owns 16 query rows. WMMA for QK^T and PV.
// ---------------------------------------------------------------------------
__global__ __launch_bounds__(128) void attention_kernel(
    const bf16_t* __restrict__ QKV, const unsigned char* __restrict__ pad,
    bf16_t* __restrict__ Ocat) {
  constexpr int LR = 72;  // 144B rows: 16B aligned, bank-skewed
  __shared__ __align__(16) bf16_t sK[64 * LR];  // [t][d]
  __shared__ __align__(16) bf16_t sV[64 * LR];  // transposed: [d][t]
  __shared__ __align__(16) bf16_t sP[64 * LR];  // [m][t]

  const int tid = threadIdx.x;
  const int lane = tid & 31, w = tid >> 5;
  const int lh = lane >> 4, lm = lane & 15;
  const int bh = blockIdx.y;
  const int b = bh >> 4, h = bh & 15;
  const int q0 = blockIdx.x * 64;
  const float scale = 0.125f;  // DK^-0.5 = 1/8

  // Q fragments straight from global (row-major in DK -> matches A layout)
  FragBF qf[2];
  {
    const uint32_t* base = (const uint32_t*)(QKV +
        (size_t)(b * Sseq + q0 + w * 16 + lm) * NQKV + h * 64);
    for (int ks = 0; ks < 2; ++ks) {
      qf[ks].q[0] = *(const uint4*)(base + ks * 16 + lh * 4);
      qf[ks].q[1] = *(const uint4*)(base + ks * 16 + 8 + lh * 4);
    }
  }

  float m_i[8], l_i[8];
  v8f accO[4];
  for (int r = 0; r < 8; ++r) { m_i[r] = -1e30f; l_i[r] = 0.f; }
  for (int dt = 0; dt < 4; ++dt) accO[dt] = {};

  for (int t0 = 0; t0 < Sseq; t0 += 64) {
    // stage K tile [t][d] via async DMA (64 rows x 128B)
    for (int i = 0; i < 4; ++i) {
      int c = tid + i * 128;
      int row = c >> 3, off8 = (c & 7) << 3;
      async_copy16(&sK[row * LR + off8],
                   QKV + (size_t)(b * Sseq + t0 + row) * NQKV + Dmod + h * 64 + off8);
    }
    // stage V transposed -> sV[d][t]: each thread takes a t-pair and 16 d's,
    // v_perm interleaves {V[t+1,d], V[t,d]} into packed dwords.
    {
      int tp = tid & 31;             // t-pair: t = 2tp, 2tp+1
      int d0 = (tid >> 5) * 16;      // 4 groups of 16 d
      const uint32_t* r0 = (const uint32_t*)(QKV +
          (size_t)(b * Sseq + t0 + 2 * tp) * NQKV + 2 * Dmod + h * 64 + d0);
      const uint32_t* r1 = (const uint32_t*)(QKV +
          (size_t)(b * Sseq + t0 + 2 * tp + 1) * NQKV + 2 * Dmod + h * 64 + d0);
      uint32_t ra[8], rb[8];
      *(uint4*)(ra)     = *(const uint4*)(r0);
      *(uint4*)(ra + 4) = *(const uint4*)(r0 + 4);
      *(uint4*)(rb)     = *(const uint4*)(r1);
      *(uint4*)(rb + 4) = *(const uint4*)(r1 + 4);
      for (int j = 0; j < 8; ++j) {
        uint32_t p0 = __builtin_amdgcn_perm(rb[j], ra[j], 0x05040100u); // {b.lo,a.lo}
        uint32_t p1 = __builtin_amdgcn_perm(rb[j], ra[j], 0x07060302u); // {b.hi,a.hi}
        *(uint32_t*)&sV[(d0 + 2 * j) * LR + 2 * tp]     = p0;
        *(uint32_t*)&sV[(d0 + 2 * j + 1) * LR + 2 * tp] = p1;
      }
    }
    wait_async0();
    __syncthreads();

    // S = Q K^T  (two k-steps of 32 over DK=64)
    v8f accS[4];
    for (int nt = 0; nt < 4; ++nt) accS[nt] = {};
    for (int ks = 0; ks < 2; ++ks)
      for (int nt = 0; nt < 4; ++nt) {
        FragBF kf;
        const uint32_t* base = (const uint32_t*)&sK[(nt * 16 + lm) * LR];
        kf.q[0] = *(const uint4*)(base + ks * 16 + lh * 4);
        kf.q[1] = *(const uint4*)(base + ks * 16 + 8 + lh * 4);
        accS[nt] = WMMA_BF16(qf[ks].v, kf.v, accS[nt]);
      }

    // key-padding mask + scale
    bool kp[4];
    for (int nt = 0; nt < 4; ++nt)
      kp[nt] = pad[b * Sseq + t0 + nt * 16 + lm] != 0;
    for (int nt = 0; nt < 4; ++nt)
      for (int r = 0; r < 8; ++r)
        accS[nt][r] = kp[nt] ? -1e30f : accS[nt][r] * scale;

    // online softmax (row = r + 8*lh, reduce over the 16 lanes of the half-wave)
    for (int r = 0; r < 8; ++r) {
      float mx = fmaxf(fmaxf(accS[0][r], accS[1][r]),
                       fmaxf(accS[2][r], accS[3][r]));
      for (int off = 1; off < 16; off <<= 1) mx = fmaxf(mx, __shfl_xor(mx, off, 32));
      float mn = fmaxf(m_i[r], mx);
      float coef = __expf(m_i[r] - mn);
      m_i[r] = mn;
      l_i[r] = l_i[r] * coef;
      for (int dt = 0; dt < 4; ++dt) accO[dt][r] = accO[dt][r] * coef;
      float ps = 0.f;
      int prow = (w * 16 + r + lh * 8) * LR;
      for (int nt = 0; nt < 4; ++nt) {
        float p = __expf(accS[nt][r] - mn);
        ps += p;
        sP[prow + nt * 16 + lm] = (bf16_t)p;  // C-layout -> A-layout via LDS
      }
      for (int off = 1; off < 16; off <<= 1) ps += __shfl_xor(ps, off, 32);
      l_i[r] += ps;
    }

    // O += P V
    for (int ks = 0; ks < 2; ++ks) {
      FragBF pf;
      const uint32_t* pb = (const uint32_t*)&sP[(w * 16 + lm) * LR];
      pf.q[0] = *(const uint4*)(pb + ks * 16 + lh * 4);
      pf.q[1] = *(const uint4*)(pb + ks * 16 + 8 + lh * 4);
      for (int dt = 0; dt < 4; ++dt) {
        FragBF vf;
        const uint32_t* vb = (const uint32_t*)&sV[(dt * 16 + lm) * LR];
        vf.q[0] = *(const uint4*)(vb + ks * 16 + lh * 4);
        vf.q[1] = *(const uint4*)(vb + ks * 16 + 8 + lh * 4);
        accO[dt] = WMMA_BF16(pf.v, vf.v, accO[dt]);
      }
    }
    __syncthreads();
  }

  // normalize, zero padded query rows, write concat-head layout [token, h*64+d]
  for (int r = 0; r < 8; ++r) {
    int qs = q0 + w * 16 + r + lh * 8;
    bool qp = pad[b * Sseq + qs] != 0;
    float inv = (!qp && l_i[r] > 0.f) ? 1.0f / l_i[r] : 0.f;
    bf16_t* orow = Ocat + (size_t)(b * Sseq + qs) * Dmod + h * 64;
    for (int dt = 0; dt < 4; ++dt)
      orow[dt * 16 + lm] = (bf16_t)(accO[dt][r] * inv);
  }
}

// ---------------------------------------------------------------------------
// Host launcher
// ---------------------------------------------------------------------------
extern "C" void kernel_launch(void* const* d_in, const int* in_sizes, int n_in,
                              void* d_out, int out_size, void* d_ws, size_t ws_size,
                              hipStream_t stream) {
  const float* src = (const float*)d_in[0];
  const unsigned char* pad = (const unsigned char*)d_in[1];
  const float* Wq = (const float*)d_in[2];
  const float* Wk = (const float*)d_in[3];
  const float* Wv = (const float*)d_in[4];
  const float* Wo = (const float*)d_in[5];
  const float* g1 = (const float*)d_in[6];
  const float* g2 = (const float*)d_in[7];
  const float* W1 = (const float*)d_in[8];
  const float* V1 = (const float*)d_in[9];
  const float* W2 = (const float*)d_in[10];

  char* ws = (char*)d_ws;
  size_t off = 0;
  auto alloc = [&](size_t bytes) -> void* {
    off = (off + 255) & ~(size_t)255;
    void* p = ws + off;
    off += bytes;
    return p;
  };
  bf16_t* Xn     = (bf16_t*)alloc((size_t)NT * Dmod * 2);
  bf16_t* QKV    = (bf16_t*)alloc((size_t)NT * NQKV * 2);
  bf16_t* Ocat   = (bf16_t*)alloc((size_t)NT * Dmod * 2);
  float*  sa     = (float*) alloc((size_t)NT * Dmod * 4);
  bf16_t* Yb     = (bf16_t*)alloc((size_t)NT * Dmod * 2);
  bf16_t* Hb     = (bf16_t*)alloc((size_t)NT * DFFf * 2);
  bf16_t* Wqkvt  = (bf16_t*)alloc((size_t)NQKV * Dmod * 2);
  bf16_t* Wot    = (bf16_t*)alloc((size_t)Dmod * Dmod * 2);
  bf16_t* W1t    = (bf16_t*)alloc((size_t)DFFf * Dmod * 2);
  bf16_t* V1t    = (bf16_t*)alloc((size_t)DFFf * Dmod * 2);
  bf16_t* W2t    = (bf16_t*)alloc((size_t)Dmod * DFFf * 2);
  (void)ws_size; (void)in_sizes; (void)n_in; (void)out_size;

  // 1) weight conversion (bf16, transposed to [N,K])
  build_wqkvt<<<(NQKV * Dmod) / 256, 256, 0, stream>>>(Wq, Wk, Wv, Wqkvt);
  transpose_f32_to_bf16<<<(Dmod * Dmod) / 256, 256, 0, stream>>>(Wo, Wot, Dmod, Dmod);
  transpose_f32_to_bf16<<<(Dmod * DFFf) / 256, 256, 0, stream>>>(W1, W1t, Dmod, DFFf);
  transpose_f32_to_bf16<<<(Dmod * DFFf) / 256, 256, 0, stream>>>(V1, V1t, Dmod, DFFf);
  transpose_f32_to_bf16<<<(DFFf * Dmod) / 256, 256, 0, stream>>>(W2, W2t, DFFf, Dmod);

  // 2) x = rmsnorm(src, g1)
  rmsnorm_bf16_kernel<<<NT, 256, 0, stream>>>(src, g1, Xn);

  // 3) QKV = x @ [Wq|Wk|Wv]   (M=8192, N=3072, K=1024) -> bf16
  gemm_bf16_kernel<2><<<dim3(NQKV / 128, NT / 128), 256, 0, stream>>>(
      Xn, Dmod, Wqkvt, nullptr, Dmod, nullptr, QKV, NQKV, NT, NQKV, Dmod);

  // 4) flash attention -> Ocat (concatenated heads, bf16)
  attention_kernel<<<dim3(Sseq / 64, Bsz * Hh), 128, 0, stream>>>(QKV, pad, Ocat);

  // 5) sa = src + Ocat @ Wo   (f32)
  gemm_bf16_kernel<1><<<dim3(Dmod / 128, NT / 128), 256, 0, stream>>>(
      Ocat, Dmod, Wot, nullptr, Dmod, src, sa, Dmod, NT, Dmod, Dmod);

  // 6) y = rmsnorm(sa, g2)
  rmsnorm_bf16_kernel<<<NT, 256, 0, stream>>>(sa, g2, Yb);

  // 7) h = silu(y@W1) * (y@V1)  (dual-B GEMM, M=8192, N=4096, K=1024) -> bf16
  gemm_bf16_kernel<3><<<dim3(DFFf / 128, NT / 128), 256, 0, stream>>>(
      Yb, Dmod, W1t, V1t, Dmod, nullptr, Hb, DFFf, NT, DFFf, Dmod);

  // 8) out = sa + h @ W2  (M=8192, N=1024, K=4096) -> f32 d_out
  gemm_bf16_kernel<1><<<dim3(Dmod / 128, NT / 128), 256, 0, stream>>>(
      Hb, DFFf, W2t, nullptr, DFFf, sa, d_out, Dmod, NT, Dmod, DFFf);
}
